// SelectiveSSM_24859270709951
// MI455X (gfx1250) — matrix operand order repaired and checked by reading, not compiled, not run on hardware
//
#include <hip/hip_runtime.h>
#include <hip/hip_bf16.h>
#include <math.h>

typedef __attribute__((ext_vector_type(16))) _Float16      v16h;
typedef __attribute__((ext_vector_type(8)))  float         v8f;
typedef __attribute__((ext_vector_type(4)))  unsigned int  v4u;
typedef __attribute__((ext_vector_type(8)))  int           v8i_;
typedef __attribute__((ext_vector_type(4)))  int           v4i_;
typedef __attribute__((address_space(3)))    float         lds_f;

#define BATCH 4
#define SEQ   1024
#define DM    1024
#define DI    2048
#define NST   16
#define RNK   64
#define NXP   128     // dbc row stride, padded from 96 to a multiple of 64

#define BS_STRIDE 65  // 64 cols + 1 dword TDM pad per row (bank-conflict free)

// ---------------------------------------------------------------------------
// Issue one TDM 2D tile load: 32 rows x 64 cols of f32 from global (row
// stride ldb elements) into LDS at byte offset lds_off, with 1-dword padding
// inserted after every 64 dwords (=> LDS row stride 65 floats).
// width_rem / krows_rem give the valid tensor extent from the tile origin;
// hardware zero-fills out-of-bounds reads.  D# layout per
// CDNA5 ISA.
// ---------------------------------------------------------------------------
__device__ static inline void tdm_load_slab(const float* gptr, unsigned lds_off,
                                            int width_rem, int krows_rem, int ldb) {
    unsigned long long ga = (unsigned long long)(const void*)gptr;
    v4u g0;
    g0[0] = 1u;                                       // count=1, user desc
    g0[1] = lds_off;                                  // lds_addr (bytes)
    g0[2] = (unsigned)(ga & 0xFFFFFFFFu);             // global_addr[31:0]
    g0[3] = (unsigned)((ga >> 32) & 0x1FFFFFFu)       // global_addr[56:32]
            | (2u << 30);                             // type = 2 ("image")
    v8i_ g1;
    g1[0] = (int)((2u << 16)                          // data_size = 4B
                  | (1u << 20)                        // pad_enable
                  | (5u << 22));                      // pad_interval: 64 dwords
                                                      // pad_amount = 0 -> 1 dword
    g1[1] = (int)(((unsigned)width_rem & 0xFFFFu) << 16);            // tdim0 lo
    g1[2] = (int)((((unsigned)width_rem >> 16) & 0xFFFFu)            // tdim0 hi
                  | (((unsigned)krows_rem & 0xFFFFu) << 16));        // tdim1 lo
    g1[3] = (int)((((unsigned)krows_rem >> 16) & 0xFFFFu)            // tdim1 hi
                  | (64u << 16));                                    // tile_dim0
    g1[4] = (int)32u;                                 // tile_dim1=32, tile_dim2=0
    g1[5] = (int)(unsigned)ldb;                       // tensor_dim0_stride lo32
    g1[6] = 0;                                        // stride0 hi16, stride1 lo16
    g1[7] = 0;                                        // stride1 hi32
    v4i_ gz = {0, 0, 0, 0};
#if __has_include(<hip/amd_detail/amd_gfx1250_TDM.h>)
    v8i_ gz8 = {0, 0, 0, 0, 0, 0, 0, 0};
    __builtin_amdgcn_tensor_load_to_lds(g0, g1, gz, gz, gz8, 0);   // clang-23 form
#else
    __builtin_amdgcn_tensor_load_to_lds(g0, g1, gz, gz, 0);        // ROCm 7.2 form
#endif
}

// ---------------------------------------------------------------------------
// WMMA GEMM: C[M,N] = A[M,K] * B[K,N], f32 in/out, f16 multiply, f32 acc.
// Requirements: M % 256 == 0, N % 64 == 0, K % 32 == 0.
// Block = 8 waves = one 256x64 macro-tile; each wave owns two 16-row tiles
// (8 WMMAs per 32-wide k-step, each LDS B fragment reused twice).
// The shared 32x64 B k-slab is double-buffered in LDS and fed by the Tensor
// Data Mover; wave 0 issues the next slab's TDM before computing the current
// one, so the DMA overlaps the WMMAs.  Nvalid (<= N) bounds the TDM read so
// the hardware zero-fills padding columns.  EPI==1: softplus(x + bias[col]).
// ---------------------------------------------------------------------------
template <int EPI>
__global__ void wmma_gemm_kernel(const float* __restrict__ A,
                                 const float* __restrict__ B,
                                 float* __restrict__ C,
                                 const float* __restrict__ bias,
                                 int M, int N, int Nvalid, int K,
                                 int lda, int ldb, int ldc) {
    __shared__ float Bs[2][32 * BS_STRIDE];

    const int lane = threadIdx.x & 31;
    const int wave = threadIdx.x >> 5;

    const int ng64 = N >> 6;
    const int tmB  = blockIdx.x / ng64;
    const int nB   = blockIdx.x % ng64;
    const int n0   = nB << 6;
    const int row0 = (tmB << 8) + (wave << 5);   // 32 rows per wave

    const int l15 = lane & 15;
    const int hi  = lane >> 4;                   // 0 or 1

    const unsigned ldsOff0 = (unsigned)(size_t)(lds_f*)&Bs[0][0];
    const unsigned ldsOff1 = (unsigned)(size_t)(lds_f*)&Bs[1][0];

    v8f acc[2][4] = {};

    if (wave == 0)                               // prologue: slab for k0 = 0
        tdm_load_slab(B + n0, ldsOff0, Nvalid - n0, K, ldb);

    int cur = 0;
    for (int k0 = 0; k0 < K; k0 += 32, cur ^= 1) {
        if (wave == 0) __builtin_amdgcn_s_wait_tensorcnt(0);
        __syncthreads();                         // Bs[cur] ready for all waves
        if (wave == 0 && k0 + 32 < K)            // prefetch next slab via TDM
            tdm_load_slab(B + (size_t)(k0 + 32) * ldb + n0,
                          cur ? ldsOff0 : ldsOff1,
                          Nvalid - n0, K - (k0 + 32), ldb);

        // ---- A fragments (global, contiguous): lanes 0-15 hold K k0+0..7 &
        //      k0+16..23, lanes 16-31 hold K k0+8..15 & k0+24..31
        v16h af[2];
#pragma unroll
        for (int rt = 0; rt < 2; ++rt) {
            const float* ap = A + (size_t)(row0 + (rt << 4) + l15) * lda
                                + k0 + (hi << 3);
            float4 a0 = *(const float4*)(ap);
            float4 a1 = *(const float4*)(ap + 4);
            float4 a2 = *(const float4*)(ap + 16);
            float4 a3 = *(const float4*)(ap + 20);
            af[rt][0]=(_Float16)a0.x;  af[rt][1]=(_Float16)a0.y;
            af[rt][2]=(_Float16)a0.z;  af[rt][3]=(_Float16)a0.w;
            af[rt][4]=(_Float16)a1.x;  af[rt][5]=(_Float16)a1.y;
            af[rt][6]=(_Float16)a1.z;  af[rt][7]=(_Float16)a1.w;
            af[rt][8]=(_Float16)a2.x;  af[rt][9]=(_Float16)a2.y;
            af[rt][10]=(_Float16)a2.z; af[rt][11]=(_Float16)a2.w;
            af[rt][12]=(_Float16)a3.x; af[rt][13]=(_Float16)a3.y;
            af[rt][14]=(_Float16)a3.z; af[rt][15]=(_Float16)a3.w;
        }

        // ---- B fragments (LDS): lanes 0-15 hold K k0+0..7 & k0+16..23, lanes
        //      16-31 K k0+8..15 & k0+24..31, column = tile*16 + (lane&15)
        const float* bsl = &Bs[cur][(hi << 3) * BS_STRIDE + l15];
#pragma unroll
        for (int tt = 0; tt < 4; ++tt) {
            v16h bf;
#pragma unroll
            for (int j = 0; j < 16; ++j)
                bf[j] = (_Float16)bsl[(j + ((j >> 3) << 3)) * BS_STRIDE + (tt << 4)];
            acc[0][tt] = __builtin_amdgcn_wmma_f32_16x16x32_f16(
                false, af[0], false, bf, (short)0, acc[0][tt], false, false);
            acc[1][tt] = __builtin_amdgcn_wmma_f32_16x16x32_f16(
                false, af[1], false, bf, (short)0, acc[1][tt], false, false);
        }
        __syncthreads();                         // done reading Bs[cur]
    }

    // ---- store: D VGPR i -> row 8*hi + i, col = lane&15
#pragma unroll
    for (int rt = 0; rt < 2; ++rt) {
#pragma unroll
        for (int tt = 0; tt < 4; ++tt) {
#pragma unroll
            for (int i = 0; i < 8; ++i) {
                const int row = row0 + (rt << 4) + (hi << 3) + i;
                const int col = n0 + (tt << 4) + l15;
                float v = acc[rt][tt][i];
                if (EPI == 1) {
                    v += bias[col];
                    v = (v > 20.f) ? v : log1pf(__expf(v));  // softplus
                }
                C[(size_t)row * ldc + col] = v;
            }
        }
    }
}

// ---------------------------------------------------------------------------
// Causal depthwise conv1d (kernel 4, left pad 3) + SiLU.
// ---------------------------------------------------------------------------
__global__ void conv_silu_kernel(const float* __restrict__ xz,
                                 const float* __restrict__ Wc,
                                 const float* __restrict__ bc,
                                 float* __restrict__ xc) {
    const int idx = blockIdx.x * blockDim.x + threadIdx.x;
    if (idx >= BATCH * SEQ * DI) return;
    const int d = idx & (DI - 1);
    const int l = (idx >> 11) & (SEQ - 1);
    const int b = idx >> 21;
    const float* xi = xz + (size_t)b * SEQ * (2 * DI);
    float s = bc[d];
#pragma unroll
    for (int t = 0; t < 4; ++t) {
        const int ls = l + t - 3;
        if (ls >= 0) s += xi[(size_t)ls * (2 * DI) + d] * Wc[d * 4 + t];
    }
    const float sig = 1.f / (1.f + __expf(-s));
    xc[idx] = s * sig;
}

// ---------------------------------------------------------------------------
// Selective scan: one thread per (b, d); 16 hidden states in registers;
// B/C rows broadcast via LDS.  Fuses y = sum_n h*C + D*x and the silu(z)
// gate, writing yz = y * silu(z).  dbc rows have stride NXP (=128).
// ---------------------------------------------------------------------------
__global__ void ssm_scan_kernel(const float* __restrict__ xz,
                                const float* __restrict__ xc,
                                const float* __restrict__ dbc,
                                const float* __restrict__ delta,
                                const float* __restrict__ A_log,
                                const float* __restrict__ Dv,
                                float* __restrict__ yz) {
    const int b = blockIdx.x >> 3;                       // DI/256 == 8
    const int d = ((blockIdx.x & 7) << 8) + threadIdx.x;

    __shared__ float sB[NST];
    __shared__ float sC[NST];

    float An[NST], h[NST];
#pragma unroll
    for (int n = 0; n < NST; ++n) {
        An[n] = -__expf(A_log[d * NST + n]);
        h[n]  = 0.f;
    }
    const float Dd = Dv[d];

    for (int l = 0; l < SEQ; ++l) {
        const size_t row = (size_t)(b * SEQ + l);
        __syncthreads();                                  // protect prev read
        if (threadIdx.x < NST)
            sB[threadIdx.x] = dbc[row * NXP + RNK + threadIdx.x];
        else if (threadIdx.x < 2 * NST)
            sC[threadIdx.x - NST] = dbc[row * NXP + RNK + NST + (threadIdx.x - NST)];
        __syncthreads();

        const float dl = delta[row * DI + d];
        const float xv = xc[row * DI + d];
        const float zv = xz[row * (2 * DI) + DI + d];

        float y = 0.f;
#pragma unroll
        for (int n = 0; n < NST; ++n) {
            const float dA = __expf(dl * An[n]);
            h[n] = dA * h[n] + (dl * sB[n]) * xv;
            y += h[n] * sC[n];
        }
        y += Dd * xv;
        const float sig = 1.f / (1.f + __expf(-zv));
        yz[row * DI + d] = y * (zv * sig);
    }
}

// ---------------------------------------------------------------------------
static inline int gemm_blocks(int M, int N) {
    return (M / 256) * (N / 64);   // one 256x64 macro-tile per block
}

extern "C" void kernel_launch(void* const* d_in, const int* in_sizes, int n_in,
                              void* d_out, int out_size, void* d_ws, size_t ws_size,
                              hipStream_t stream) {
    const float* x       = (const float*)d_in[0];
    const float* W_in    = (const float*)d_in[1];
    const float* W_conv  = (const float*)d_in[2];
    const float* b_conv  = (const float*)d_in[3];
    const float* W_xproj = (const float*)d_in[4];
    const float* W_dt    = (const float*)d_in[5];
    const float* b_dt    = (const float*)d_in[6];
    const float* A_log   = (const float*)d_in[7];
    const float* Dv      = (const float*)d_in[8];
    const float* W_out   = (const float*)d_in[9];
    float* out = (float*)d_out;

    const int M = BATCH * SEQ;  // 4096 token rows

    float* ws    = (float*)d_ws;
    float* xz    = ws;                          // M * 2*DI
    float* xc    = xz    + (size_t)M * 2 * DI;  // M * DI
    float* dbc   = xc    + (size_t)M * DI;      // M * NXP (padded to 128)
    float* delta = dbc   + (size_t)M * NXP;     // M * DI
    float* yz    = delta + (size_t)M * DI;      // M * DI   (~163 MB total, fits L2)

    // 1) xz = x @ W_in                       [4096,1024]x[1024,4096]
    wmma_gemm_kernel<0><<<gemm_blocks(M, 2 * DI), 256, 0, stream>>>(
        x, W_in, xz, nullptr, M, 2 * DI, 2 * DI, DM, DM, 2 * DI, 2 * DI);

    // 2) xc = silu(causal_dwconv(xi))
    conv_silu_kernel<<<(BATCH * SEQ * DI + 255) / 256, 256, 0, stream>>>(
        xz, W_conv, b_conv, xc);

    // 3) dbc = xc @ W_xproj                  [4096,2048]x[2048,96], N padded
    //    to 128; TDM zero-fills cols 96..127 (Nvalid=96), stored with ldc=128.
    wmma_gemm_kernel<0><<<gemm_blocks(M, NXP), 256, 0, stream>>>(
        xc, W_xproj, dbc, nullptr, M, NXP, 96, DI, DI, 96, NXP);

    // 4) delta = softplus(dbc[:, :64] @ W_dt + b_dt)   (A view: lda = 128)
    wmma_gemm_kernel<1><<<gemm_blocks(M, DI), 256, 0, stream>>>(
        dbc, W_dt, delta, b_dt, M, DI, DI, RNK, NXP, DI, DI);

    // 5) selective scan + silu(z) gate -> yz
    ssm_scan_kernel<<<BATCH * (DI / 256), 256, 0, stream>>>(
        xz, xc, dbc, delta, A_log, Dv, yz);

    // 6) out = yz @ W_out                    [4096,2048]x[2048,1024]
    wmma_gemm_kernel<0><<<gemm_blocks(M, DM), 256, 0, stream>>>(
        yz, W_out, out, nullptr, M, DM, DM, DI, DI, DM, DM);
}
